// MultiHeadAttentionBlock_1185410974391
// MI455X (gfx1250) — compile-verified, matrix operand order checked by reading, not run on hardware
//
#include <hip/hip_runtime.h>

// ---------------------------------------------------------------------------
// MHA block for MI455X (gfx1250, wave32, WMMA).
//
// All GEMMs run on v_wmma_f32_16x16x32_bf16 (f32 accumulate).  Every GEMM in
// this block is "NT" (K contiguous for both operands), and since a WMMA dot
// product is invariant under a shared permutation of K slots, each lane just
// loads 16 contiguous bf16 K-values for its row/col:
//    lane L: row/col = base + (L & 15),  K chunk = k0 + (L >> 4) * 16
// C/D fragment layout (ISA 7.12.2): VGPR r, lane l -> M = r + (l>>4)*8,
// N = l & 15.
// ---------------------------------------------------------------------------

typedef __bf16 bf16_t;
typedef __attribute__((ext_vector_type(16))) __bf16 v16bf;
typedef __attribute__((ext_vector_type(8)))  float  v8f;
typedef __attribute__((ext_vector_type(16))) float  v16f;

#define B_   2
#define S_   2048
#define D_   1024
#define H_   16
#define HD_  64
#define M_   (B_ * S_)   // 4096 rows of x
#define N3_  (3 * D_)    // 3072 qkv cols

static __device__ inline v8f wmma_bf16(v16bf a, v16bf b, v8f c) {
    return __builtin_amdgcn_wmma_f32_16x16x32_bf16(
        /*neg_a=*/false, a, /*neg_b=*/false, b,
        /*c_mod=*/(short)0, c, /*reuse_a=*/false, /*reuse_b=*/false);
}

// ---------------------------------------------------------------------------
// fp32 -> bf16 conversion (grid-stride not needed; exact launch)
// ---------------------------------------------------------------------------
__global__ void cvt_bf16_kernel(const float* __restrict__ in,
                                bf16_t* __restrict__ out, int n) {
    int i = blockIdx.x * blockDim.x + threadIdx.x;
    if (i < n) out[i] = (bf16_t)in[i];
}

// ---------------------------------------------------------------------------
// QKV GEMM:  qkv[m,n] = x[m,:] . W_w[n,:] + W_b[n]      (M=4096, N=3072, K=1024)
// Epilogue scatters to q[b,h,s,hd], k[b,h,s,hd], vT[b,h,hd,s] in bf16.
// Block = 256 thr = 8 waves (2x4), block tile 64(M) x 128(N), wave tile 32x32.
// ---------------------------------------------------------------------------
__global__ void __launch_bounds__(256) qkv_gemm_kernel(
    const bf16_t* __restrict__ xh, const bf16_t* __restrict__ wh,
    const float* __restrict__ wb,
    bf16_t* __restrict__ qh, bf16_t* __restrict__ kh, bf16_t* __restrict__ vth)
{
    const int lane = threadIdx.x & 31;
    const int wave = threadIdx.x >> 5;
    const int wm = wave >> 2, wn = wave & 3;
    const int m_blk = blockIdx.y * 64  + wm * 32;
    const int n_blk = blockIdx.x * 128 + wn * 32;
    const int rsel = lane & 15;
    const int ksel = (lane >> 4) * 16;

    const bf16_t* a0p = xh + (size_t)(m_blk + rsel) * D_ + ksel;
    const bf16_t* a1p = a0p + (size_t)16 * D_;
    const bf16_t* b0p = wh + (size_t)(n_blk + rsel) * D_ + ksel;
    const bf16_t* b1p = b0p + (size_t)16 * D_;

    v8f acc[2][2] = {};
    for (int k0 = 0; k0 < D_; k0 += 32) {
        v16bf a0 = *(const v16bf*)(a0p + k0);
        v16bf a1 = *(const v16bf*)(a1p + k0);
        v16bf b0 = *(const v16bf*)(b0p + k0);
        v16bf b1 = *(const v16bf*)(b1p + k0);
        acc[0][0] = wmma_bf16(a0, b0, acc[0][0]);
        acc[0][1] = wmma_bf16(a0, b1, acc[0][1]);
        acc[1][0] = wmma_bf16(a1, b0, acc[1][0]);
        acc[1][1] = wmma_bf16(a1, b1, acc[1][1]);
    }

    #pragma unroll
    for (int i = 0; i < 2; ++i)
    #pragma unroll
    for (int j = 0; j < 2; ++j) {
        #pragma unroll
        for (int r = 0; r < 8; ++r) {
            int m = m_blk + i * 16 + (lane >> 4) * 8 + r;
            int n = n_blk + j * 16 + (lane & 15);
            float val = acc[i][j][r] + wb[n];
            int bq = m >> 11;            // m / S_
            int srow = m & (S_ - 1);
            int which = n >> 10;         // 0:q 1:k 2:v
            int col = n & (D_ - 1);
            int h = col >> 6;
            int hc = col & (HD_ - 1);
            bf16_t bv = (bf16_t)val;
            size_t bh = (size_t)(bq * H_ + h);
            if (which == 0)       qh[(bh * S_ + srow) * HD_ + hc] = bv;
            else if (which == 1)  kh[(bh * S_ + srow) * HD_ + hc] = bv;
            else                  vth[(bh * HD_ + hc) * S_ + srow] = bv;
        }
    }
}

// ---------------------------------------------------------------------------
// Scores: attn[bh,q,k] = (q[bh,q,:] . k[bh,k,:]) / sqrt(D)   (per bh: 2048x2048x64)
// Written fp32 directly into the d_out attn region.
// ---------------------------------------------------------------------------
__global__ void __launch_bounds__(256) scores_kernel(
    const bf16_t* __restrict__ qh, const bf16_t* __restrict__ kh,
    float* __restrict__ attn)
{
    const int bh = blockIdx.z;
    const int lane = threadIdx.x & 31;
    const int wave = threadIdx.x >> 5;
    const int wm = wave >> 2, wn = wave & 3;
    const int m_blk = blockIdx.y * 64  + wm * 32;
    const int n_blk = blockIdx.x * 128 + wn * 32;
    const int rsel = lane & 15;
    const int ksel = (lane >> 4) * 16;

    const bf16_t* a0p = qh + (size_t)bh * S_ * HD_ + (size_t)(m_blk + rsel) * HD_ + ksel;
    const bf16_t* a1p = a0p + (size_t)16 * HD_;
    const bf16_t* b0p = kh + (size_t)bh * S_ * HD_ + (size_t)(n_blk + rsel) * HD_ + ksel;
    const bf16_t* b1p = b0p + (size_t)16 * HD_;

    v8f acc[2][2] = {};
    #pragma unroll
    for (int k0 = 0; k0 < HD_; k0 += 32) {
        v16bf a0 = *(const v16bf*)(a0p + k0);
        v16bf a1 = *(const v16bf*)(a1p + k0);
        v16bf b0 = *(const v16bf*)(b0p + k0);
        v16bf b1 = *(const v16bf*)(b1p + k0);
        acc[0][0] = wmma_bf16(a0, b0, acc[0][0]);
        acc[0][1] = wmma_bf16(a0, b1, acc[0][1]);
        acc[1][0] = wmma_bf16(a1, b0, acc[1][0]);
        acc[1][1] = wmma_bf16(a1, b1, acc[1][1]);
    }

    float* ob = attn + (size_t)bh * S_ * S_;
    const float scale = 0.03125f;  // 1/sqrt(1024)
    #pragma unroll
    for (int i = 0; i < 2; ++i)
    #pragma unroll
    for (int j = 0; j < 2; ++j) {
        #pragma unroll
        for (int r = 0; r < 8; ++r) {
            int m = m_blk + i * 16 + (lane >> 4) * 8 + r;
            int n = n_blk + j * 16 + (lane & 15);
            ob[(size_t)m * S_ + n] = acc[i][j][r] * scale;
        }
    }
}

// ---------------------------------------------------------------------------
// Row softmax over 2048 elems, in place in d_out. One block per row.
// ---------------------------------------------------------------------------
__global__ void __launch_bounds__(256) softmax_kernel(float* __restrict__ attn)
{
    __shared__ float red[256];
    float* p = attn + (size_t)blockIdx.x * S_;
    const int t = threadIdx.x;

    float vals[8];
    float m = -3.4e38f;
    #pragma unroll
    for (int i = 0; i < 8; ++i) { vals[i] = p[t + i * 256]; m = fmaxf(m, vals[i]); }
    red[t] = m; __syncthreads();
    for (int s = 128; s > 0; s >>= 1) {
        if (t < s) red[t] = fmaxf(red[t], red[t + s]);
        __syncthreads();
    }
    m = red[0]; __syncthreads();

    float sum = 0.0f;
    #pragma unroll
    for (int i = 0; i < 8; ++i) { vals[i] = __expf(vals[i] - m); sum += vals[i]; }
    red[t] = sum; __syncthreads();
    for (int s = 128; s > 0; s >>= 1) {
        if (t < s) red[t] += red[t + s];
        __syncthreads();
    }
    float inv = 1.0f / red[0];
    #pragma unroll
    for (int i = 0; i < 8; ++i) p[t + i * 256] = vals[i] * inv;
}

// ---------------------------------------------------------------------------
// AV: o[bh,q,n] = sum_k attn[bh,q,k] * vT[bh,n,k]   (per bh: 2048 x 64 x 2048)
// attn read fp32 from d_out, converted to bf16 in-register.
// Block = 8 waves (4x2), tile 128(M) x 64(N). Output merged to [b,s,d] bf16.
// ---------------------------------------------------------------------------
__global__ void __launch_bounds__(256) av_kernel(
    const float* __restrict__ attn, const bf16_t* __restrict__ vth,
    bf16_t* __restrict__ oh)
{
    const int bh = blockIdx.z;
    const int bq = bh >> 4, h = bh & 15;
    const int lane = threadIdx.x & 31;
    const int wave = threadIdx.x >> 5;
    const int wm = wave >> 1, wn = wave & 1;
    const int m_blk = blockIdx.y * 128 + wm * 32;
    const int n_blk = wn * 32;
    const int rsel = lane & 15;
    const int ksel = (lane >> 4) * 16;

    const float* a0p = attn + (size_t)bh * S_ * S_ + (size_t)(m_blk + rsel) * S_ + ksel;
    const float* a1p = a0p + (size_t)16 * S_;
    const bf16_t* b0p = vth + (size_t)bh * HD_ * S_ + (size_t)(n_blk + rsel) * S_ + ksel;
    const bf16_t* b1p = b0p + (size_t)16 * S_;

    v8f acc[2][2] = {};
    for (int k0 = 0; k0 < S_; k0 += 32) {
        v16f af0 = *(const v16f*)(a0p + k0);
        v16f af1 = *(const v16f*)(a1p + k0);
        v16bf a0, a1;
        #pragma unroll
        for (int i = 0; i < 16; ++i) { a0[i] = (bf16_t)af0[i]; a1[i] = (bf16_t)af1[i]; }
        v16bf b0 = *(const v16bf*)(b0p + k0);
        v16bf b1 = *(const v16bf*)(b1p + k0);
        acc[0][0] = wmma_bf16(a0, b0, acc[0][0]);
        acc[0][1] = wmma_bf16(a0, b1, acc[0][1]);
        acc[1][0] = wmma_bf16(a1, b0, acc[1][0]);
        acc[1][1] = wmma_bf16(a1, b1, acc[1][1]);
    }

    #pragma unroll
    for (int i = 0; i < 2; ++i)
    #pragma unroll
    for (int j = 0; j < 2; ++j) {
        #pragma unroll
        for (int r = 0; r < 8; ++r) {
            int q = m_blk + i * 16 + (lane >> 4) * 8 + r;
            int n = n_blk + j * 16 + (lane & 15);
            oh[((size_t)bq * S_ + q) * D_ + h * HD_ + n] = (bf16_t)acc[i][j][r];
        }
    }
}

// ---------------------------------------------------------------------------
// Projection + bias + residual: y[m,n] = x[m,n] + proj_b[n] + o[m,:] . proj_w[n,:]
// (M=4096, N=1024, K=1024).  y written fp32 into d_out (pre-LayerNorm).
// ---------------------------------------------------------------------------
__global__ void __launch_bounds__(256) proj_kernel(
    const bf16_t* __restrict__ oh, const bf16_t* __restrict__ wph,
    const float* __restrict__ pb, const float* __restrict__ x,
    float* __restrict__ y)
{
    const int lane = threadIdx.x & 31;
    const int wave = threadIdx.x >> 5;
    const int wm = wave >> 2, wn = wave & 3;
    const int m_blk = blockIdx.y * 64  + wm * 32;
    const int n_blk = blockIdx.x * 128 + wn * 32;
    const int rsel = lane & 15;
    const int ksel = (lane >> 4) * 16;

    const bf16_t* a0p = oh + (size_t)(m_blk + rsel) * D_ + ksel;
    const bf16_t* a1p = a0p + (size_t)16 * D_;
    const bf16_t* b0p = wph + (size_t)(n_blk + rsel) * D_ + ksel;
    const bf16_t* b1p = b0p + (size_t)16 * D_;

    v8f acc[2][2] = {};
    for (int k0 = 0; k0 < D_; k0 += 32) {
        v16bf a0 = *(const v16bf*)(a0p + k0);
        v16bf a1 = *(const v16bf*)(a1p + k0);
        v16bf b0 = *(const v16bf*)(b0p + k0);
        v16bf b1 = *(const v16bf*)(b1p + k0);
        acc[0][0] = wmma_bf16(a0, b0, acc[0][0]);
        acc[0][1] = wmma_bf16(a0, b1, acc[0][1]);
        acc[1][0] = wmma_bf16(a1, b0, acc[1][0]);
        acc[1][1] = wmma_bf16(a1, b1, acc[1][1]);
    }

    #pragma unroll
    for (int i = 0; i < 2; ++i)
    #pragma unroll
    for (int j = 0; j < 2; ++j) {
        #pragma unroll
        for (int r = 0; r < 8; ++r) {
            int m = m_blk + i * 16 + (lane >> 4) * 8 + r;
            int n = n_blk + j * 16 + (lane & 15);
            size_t idx = (size_t)m * D_ + n;
            y[idx] = acc[i][j][r] + pb[n] + x[idx];
        }
    }
}

// ---------------------------------------------------------------------------
// LayerNorm over D=1024, in place in d_out. One block per row.
// ---------------------------------------------------------------------------
__global__ void __launch_bounds__(256) ln_kernel(
    float* __restrict__ y, const float* __restrict__ gamma,
    const float* __restrict__ beta)
{
    __shared__ float rs[256];
    __shared__ float rs2[256];
    float* p = y + (size_t)blockIdx.x * D_;
    const int t = threadIdx.x;

    float v[4];
    float s = 0.0f, s2 = 0.0f;
    #pragma unroll
    for (int i = 0; i < 4; ++i) {
        v[i] = p[t + i * 256];
        s += v[i]; s2 += v[i] * v[i];
    }
    rs[t] = s; rs2[t] = s2; __syncthreads();
    for (int k = 128; k > 0; k >>= 1) {
        if (t < k) { rs[t] += rs[t + k]; rs2[t] += rs2[t + k]; }
        __syncthreads();
    }
    float mu  = rs[0] * (1.0f / D_);
    float var = rs2[0] * (1.0f / D_) - mu * mu;
    float inv = rsqrtf(var + 1e-5f);
    #pragma unroll
    for (int i = 0; i < 4; ++i) {
        int c = t + i * 256;
        p[c] = (v[i] - mu) * inv * gamma[c] + beta[c];
    }
}

// ---------------------------------------------------------------------------
extern "C" void kernel_launch(void* const* d_in, const int* in_sizes, int n_in,
                              void* d_out, int out_size, void* d_ws, size_t ws_size,
                              hipStream_t stream) {
    (void)in_sizes; (void)n_in; (void)out_size; (void)ws_size;

    const float* x      = (const float*)d_in[0];
    const float* W_w    = (const float*)d_in[1];
    const float* W_b    = (const float*)d_in[2];
    const float* proj_w = (const float*)d_in[3];
    const float* proj_b = (const float*)d_in[4];
    const float* gamma  = (const float*)d_in[5];
    const float* beta   = (const float*)d_in[6];

    float* y_norm = (float*)d_out;                       // [b,s,d] = 4,194,304 f32
    float* attn   = y_norm + (size_t)M_ * D_;            // [b,h,s,s] = 134,217,728 f32

    // Workspace layout (48 MB total, all 256B-aligned)
    char* ws = (char*)d_ws;
    bf16_t* xh    = (bf16_t*)(ws);                 //  8 MB  x bf16 [4096,1024]
    bf16_t* wqkvh = (bf16_t*)(ws +  8388608);      //  6 MB  W_w bf16 [3072,1024]
    bf16_t* wph   = (bf16_t*)(ws + 14680064);      //  2 MB  proj_w bf16 [1024,1024]
    bf16_t* qh    = (bf16_t*)(ws + 16777216);      //  8 MB  q  [b,h,s,hd]
    bf16_t* kh    = (bf16_t*)(ws + 25165824);      //  8 MB  k  [b,h,s,hd]
    bf16_t* vth   = (bf16_t*)(ws + 33554432);      //  8 MB  vT [b,h,hd,s]
    bf16_t* oh    = (bf16_t*)(ws + 41943040);      //  8 MB  o  [b,s,d]

    // 1) fp32 -> bf16 conversions
    cvt_bf16_kernel<<<(M_ * D_) / 256, 256, 0, stream>>>(x, xh, M_ * D_);
    cvt_bf16_kernel<<<(N3_ * D_) / 256, 256, 0, stream>>>(W_w, wqkvh, N3_ * D_);
    cvt_bf16_kernel<<<(D_ * D_) / 256, 256, 0, stream>>>(proj_w, wph, D_ * D_);

    // 2) QKV GEMM (4096 x 3072 x 1024) -> q/k/vT
    qkv_gemm_kernel<<<dim3(N3_ / 128, M_ / 64), 256, 0, stream>>>(
        xh, wqkvh, W_b, qh, kh, vth);

    // 3) Scores per (b,h): 2048 x 2048 x 64, scaled
    scores_kernel<<<dim3(S_ / 128, S_ / 64, B_ * H_), 256, 0, stream>>>(
        qh, kh, attn);

    // 4) Softmax over rows (in place in d_out)
    softmax_kernel<<<dim3(B_ * H_ * S_), 256, 0, stream>>>(attn);

    // 5) AV per (b,h): 2048 x 64 x 2048 -> o bf16 [b,s,d]
    av_kernel<<<dim3(1, S_ / 128, B_ * H_), 256, 0, stream>>>(attn, vth, oh);

    // 6) Projection + bias + residual (4096 x 1024 x 1024) -> y fp32 in d_out
    proj_kernel<<<dim3(D_ / 128, M_ / 64), 256, 0, stream>>>(
        oh, wph, proj_b, x, y_norm);

    // 7) LayerNorm in place
    ln_kernel<<<dim3(M_), 256, 0, stream>>>(y_norm, gamma, beta);
}
